// RangeAwareGNN_25022479467020
// MI455X (gfx1250) — matrix operand order repaired and checked by reading, not compiled, not run on hardware
//
#include <hip/hip_runtime.h>
#include <hip/hip_bf16.h>

typedef __attribute__((ext_vector_type(16))) _Float16 v16h;
typedef __attribute__((ext_vector_type(8)))  _Float16 v8h;
typedef __attribute__((ext_vector_type(8)))  float    v8f;

#define N_NODES 50000
#define N_EDGES 640000
#define F_IN    64
#define HDIM    128
#define NGRAPH  256
#define NEG_SLOPE 0.2f
#define BN_EPS  1e-5f

// ---------- helpers ----------
__device__ __forceinline__ unsigned f2ord(float f) {
    unsigned u = __float_as_uint(f);
    return (u & 0x80000000u) ? ~u : (u | 0x80000000u);
}
__device__ __forceinline__ float ord2f(unsigned u) {
    return __uint_as_float((u & 0x80000000u) ? (u ^ 0x80000000u) : ~u);
}

// ---------- f32 -> f16 convert ----------
__global__ void cvt_f16_k(const float* __restrict__ in, _Float16* __restrict__ out, long total) {
    long t = (long)blockIdx.x * blockDim.x + threadIdx.x;
    if (t < total) out[t] = (_Float16)in[t];
}

// ---------- pack W[K x 128] into per-lane WMMA B-matrix layout ----------
// linear index t = ((ct*nchunks + c)*32 + lane)*16 + e
// value = W[32*c + 16*(lane>>4) + e][ct*16 + (lane&15)]
__global__ void pack_w_k(const float* __restrict__ W, _Float16* __restrict__ wp, int nchunks) {
    int t = blockIdx.x * blockDim.x + threadIdx.x;
    int total = nchunks * 8 * 512;
    if (t >= total) return;
    int e    = t & 15;
    int lane = (t >> 4) & 31;
    int rest = t >> 9;                 // ct*nchunks + c
    int c  = rest % nchunks;
    int ct = rest / nchunks;
    int k = 32 * c + 16 * (lane >> 4) + e;
    int n = ct * 16 + (lane & 15);
    wp[t] = (_Float16)W[k * HDIM + n];
}

// ---------- WMMA GEMM: D[nrows x 128] = A[nrows x K](f16,row-major) @ Bpack ----------
// Block = 256 threads = 8 waves. Block tile: 128 rows x 128 cols.
// Wave w owns rows [blk*128 + w*16, +16); keeps 8 f32 accumulators (one per col tile).
// Packed B (K x 128, WMMA B layout) is staged once into LDS and read back via ds loads.
template <int NCHUNKS>
__global__ void wmma_gemm_k(const _Float16* __restrict__ A,
                            const _Float16* __restrict__ Bp,
                            float* __restrict__ D, int lda, int nrows) {
    __shared__ _Float16 lds_b[NCHUNKS * 8 * 512];

    // cooperative stage of packed B into LDS (16B vectors)
    const int nvec = NCHUNKS * 8 * 512 / 8;      // halves / 8 per uint4
    for (int i = threadIdx.x; i < nvec; i += 256)
        ((uint4*)lds_b)[i] = ((const uint4*)Bp)[i];
    __syncthreads();

    const int lane = threadIdx.x & 31;
    const int w    = threadIdx.x >> 5;            // wave = row sub-tile 0..7
    const int g    = lane >> 4;
    const int l15  = lane & 15;
    const long row0 = (long)blockIdx.x * 128 + w * 16;

    // clamp A row so OOB rows read row (nrows-1); their stores are predicated off
    const long arow_i = (row0 + l15 < nrows) ? (row0 + l15) : (long)(nrows - 1);
    const _Float16* arow = A + (size_t)arow_i * lda;

    v8f acc[8] = {};
#pragma unroll
    for (int c = 0; c < NCHUNKS; ++c) {
        union { v16h v; v8h h[2]; } a;
        const int k0 = 32 * c;
        // A 16x32 f16 layout: elem e (0..15) -> K = k0+16g+e
        a.h[0] = *(const v8h*)(arow + k0 + 16 * g);
        a.h[1] = *(const v8h*)(arow + k0 + 16 * g + 8);
#pragma unroll
        for (int ct = 0; ct < 8; ++ct) {
            union { v16h v; v8h h[2]; } b;
            const _Float16* bb = lds_b + ((size_t)(ct * NCHUNKS + c) * 32 + lane) * 16;
            b.h[0] = *(const v8h*)bb;
            b.h[1] = *(const v8h*)(bb + 8);
            acc[ct] = __builtin_amdgcn_wmma_f32_16x16x32_f16(false, a.v, false, b.v,
                                                             (short)0, acc[ct], false, false);
        }
    }
    // C/D layout: VGPR v -> row = row0 + 8*g + v ; col = ct*16 + (lane&15)
#pragma unroll
    for (int ct = 0; ct < 8; ++ct) {
#pragma unroll
        for (int v = 0; v < 8; ++v) {
            long row = row0 + 8 * g + v;
            if (row < nrows) D[(size_t)row * HDIM + ct * 16 + l15] = acc[ct][v];
        }
    }
}

// ---------- per-node attention scores: s1=h.a_src, s2=h.a_dst (one wave per node) ----------
__global__ void row_dot2_k(const float* __restrict__ h,
                           const float* __restrict__ a1, const float* __restrict__ a2,
                           float* __restrict__ s1, float* __restrict__ s2, int n) {
    int wid  = blockIdx.x * (blockDim.x >> 5) + (threadIdx.x >> 5);
    int lane = threadIdx.x & 31;
    if (wid >= n) return;
    float4 hv = ((const float4*)(h + (size_t)wid * HDIM))[lane];
    float4 w1 = ((const float4*)a1)[lane];
    float4 w2 = ((const float4*)a2)[lane];
    float d1 = hv.x * w1.x + hv.y * w1.y + hv.z * w1.z + hv.w * w1.w;
    float d2 = hv.x * w2.x + hv.y * w2.y + hv.z * w2.z + hv.w * w2.w;
#pragma unroll
    for (int off = 16; off; off >>= 1) {
        d1 += __shfl_xor(d1, off, 32);
        d2 += __shfl_xor(d2, off, 32);
    }
    if (lane == 0) { s1[wid] = d1; s2[wid] = d2; }
}

// ---------- init segment buffers ----------
__global__ void init_seg_k(unsigned* __restrict__ mm, float* __restrict__ den,
                           float* __restrict__ agg, int n, long total128) {
    long t = (long)blockIdx.x * blockDim.x + threadIdx.x;
    if (t < total128) agg[t] = 0.f;
    if (t < n) { mm[t] = 0x007FFFFFu; /* f2ord(-inf) */ den[t] = 0.f; }
}

// ---------- edge pass 1: segment max ----------
__global__ void edge_max_k(const int* __restrict__ ei, const float* __restrict__ s1,
                           const float* __restrict__ s2, unsigned* __restrict__ mm,
                           int E, int n) {
    long i = (long)blockIdx.x * blockDim.x + threadIdx.x;
    long tot = (long)E + n;
    if (i >= tot) return;
    int src, dst;
    if (i < E) { src = ei[i]; dst = ei[E + i]; } else { src = dst = (int)(i - E); }
    float e = s1[src] + s2[dst];
    e = (e < 0.f) ? NEG_SLOPE * e : e;
    atomicMax(&mm[dst], f2ord(e));
}

// ---------- edge pass 2: exp + segment sum ----------
__global__ void edge_expsum_k(const int* __restrict__ ei, const float* __restrict__ s1,
                              const float* __restrict__ s2, const unsigned* __restrict__ mm,
                              float* __restrict__ den, float* __restrict__ coef,
                              int E, int n) {
    long i = (long)blockIdx.x * blockDim.x + threadIdx.x;
    long tot = (long)E + n;
    if (i >= tot) return;
    int src, dst;
    if (i < E) { src = ei[i]; dst = ei[E + i]; } else { src = dst = (int)(i - E); }
    float e = s1[src] + s2[dst];
    e = (e < 0.f) ? NEG_SLOPE * e : e;
    float ex = __expf(e - ord2f(mm[dst]));
    coef[i] = ex;
    atomicAdd(&den[dst], ex);
}

// ---------- edge pass 3: weighted scatter-add (thread = channel) ----------
#define SC_EDGES 8
__global__ void scatter_k(const int* __restrict__ ei, const float* __restrict__ coef,
                          const float* __restrict__ den, const float* __restrict__ h,
                          float* __restrict__ agg, int E, int n) {
    int tid = threadIdx.x;                    // 0..127
    long base = (long)blockIdx.x * SC_EDGES;
    long tot = (long)E + n;
#pragma unroll
    for (int j = 0; j < SC_EDGES; ++j) {
        long e = base + j;
        if (e >= tot) return;
        int src, dst;
        if (e < E) { src = ei[e]; dst = ei[E + e]; } else { src = dst = (int)(e - E); }
        // hide the h[src] pointer-chase of the NEXT edge behind this edge's atomics
        long e2 = e + 1;
        if (e2 < tot) {
            int s2 = (e2 < E) ? ei[e2] : (int)(e2 - E);
            __builtin_prefetch(&h[(size_t)s2 * HDIM + tid], 0, 1);   // global_prefetch_b8
        }
        float c = coef[e] / den[dst];
        atomicAdd(&agg[(size_t)dst * HDIM + tid], c * h[(size_t)src * HDIM + tid]);
    }
}

// ---------- bias + BN + ReLU, write f32 and f16 ----------
__global__ void postact_k(const float* __restrict__ agg, const float* __restrict__ bias,
                          const float* __restrict__ gm, const float* __restrict__ bt,
                          const float* __restrict__ mu, const float* __restrict__ vr,
                          float* __restrict__ hout, _Float16* __restrict__ h16, long total) {
    long t = (long)blockIdx.x * blockDim.x + threadIdx.x;
    if (t >= total) return;
    int c = (int)(t & (HDIM - 1));
    float v = agg[t] + bias[c];
    v = (v - mu[c]) * rsqrtf(vr[c] + BN_EPS) * gm[c] + bt[c];
    v = fmaxf(v, 0.f);
    hout[t] = v;
    h16[t] = (_Float16)v;
}

// ---------- pooling ----------
__global__ void zero_pool_k(float* __restrict__ pool, float* __restrict__ cnt) {
    int t = blockIdx.x * blockDim.x + threadIdx.x;
    if (t < NGRAPH * HDIM) pool[t] = 0.f;
    if (t < NGRAPH) cnt[t] = 0.f;
}
#define PL_NODES 32
__global__ void pool_sum_k(const int* __restrict__ batch, const float* __restrict__ h,
                           float* __restrict__ pool, float* __restrict__ cnt, int n) {
    int tid = threadIdx.x;                    // 0..127
    long base = (long)blockIdx.x * PL_NODES;
    for (int j = 0; j < PL_NODES; ++j) {
        long i = base + j;
        if (i >= n) return;
        int b = batch[i];
        atomicAdd(&pool[(size_t)b * HDIM + tid], h[i * HDIM + tid]);
        if (tid == 0) atomicAdd(&cnt[b], 1.0f);
    }
}
__global__ void pool_div_k(float* __restrict__ pool, const float* __restrict__ cnt,
                           _Float16* __restrict__ p16) {
    int t = blockIdx.x * blockDim.x + threadIdx.x;
    if (t >= NGRAPH * HDIM) return;
    float p = pool[t] / fmaxf(cnt[t >> 7], 1.0f);
    pool[t] = p;
    p16[t] = (_Float16)p;
}

// ---------- MLP head ----------
__global__ void hid_act_k(float* __restrict__ hid, const float* __restrict__ lb1) {
    int t = blockIdx.x * blockDim.x + threadIdx.x;
    if (t >= NGRAPH * HDIM) return;
    hid[t] = fmaxf(hid[t] + lb1[t & (HDIM - 1)], 0.f);
}
__global__ void final_head_k(const float* __restrict__ hid, const float* __restrict__ lw2,
                             const float* __restrict__ lb2, float* __restrict__ out) {
    int g = blockIdx.x * blockDim.x + threadIdx.x;
    if (g >= NGRAPH) return;
    float acc = lb2[0];
#pragma unroll 8
    for (int c = 0; c < HDIM; ++c) acc += hid[(size_t)g * HDIM + c] * lw2[c];
    out[g] = acc;
}

// ===================================================================
extern "C" void kernel_launch(void* const* d_in, const int* in_sizes, int n_in,
                              void* d_out, int out_size, void* d_ws, size_t ws_size,
                              hipStream_t stream) {
    const float* X   = (const float*)d_in[0];
    const int*   EI  = (const int*)d_in[1];
    const int*   BATCH = (const int*)d_in[2];
    const float* W1 = (const float*)d_in[3],  *as1 = (const float*)d_in[4],
               * ad1 = (const float*)d_in[5], *b1  = (const float*)d_in[6];
    const float* W2 = (const float*)d_in[7],  *as2 = (const float*)d_in[8],
               * ad2 = (const float*)d_in[9], *b2  = (const float*)d_in[10];
    const float* W3 = (const float*)d_in[11], *as3 = (const float*)d_in[12],
               * ad3 = (const float*)d_in[13], *b3 = (const float*)d_in[14];
    const float* g1 = (const float*)d_in[15], *be1 = (const float*)d_in[16],
               * m1 = (const float*)d_in[17], *v1  = (const float*)d_in[18];
    const float* g2 = (const float*)d_in[19], *be2 = (const float*)d_in[20],
               * m2 = (const float*)d_in[21], *v2  = (const float*)d_in[22];
    const float* g3 = (const float*)d_in[23], *be3 = (const float*)d_in[24],
               * m3 = (const float*)d_in[25], *v3  = (const float*)d_in[26];
    const float* lw1 = (const float*)d_in[27], *lb1 = (const float*)d_in[28];
    const float* lw2 = (const float*)d_in[29], *lb2 = (const float*)d_in[30];

    // workspace bump allocator (256B aligned regions)
    char* wsp = (char*)d_ws;
    auto take = [&](size_t bytes) {
        char* p = wsp; wsp += (bytes + 255) & ~(size_t)255; return (void*)p;
    };
    _Float16* X16 = (_Float16*)take((size_t)N_NODES * HDIM * 2);
    float*    H   = (float*)take((size_t)N_NODES * HDIM * 4);
    float*    AGG = (float*)take((size_t)N_NODES * HDIM * 4);
    float*    S1  = (float*)take((size_t)N_NODES * 4);
    float*    S2  = (float*)take((size_t)N_NODES * 4);
    unsigned* MM  = (unsigned*)take((size_t)N_NODES * 4);
    float*    DEN = (float*)take((size_t)N_NODES * 4);
    float*    COEF = (float*)take((size_t)(N_EDGES + N_NODES) * 4);
    _Float16* WP  = (_Float16*)take((size_t)HDIM * HDIM * 2);
    float*    POOL = (float*)take((size_t)NGRAPH * HDIM * 4);
    float*    CNT  = (float*)take((size_t)NGRAPH * 4);
    _Float16* P16  = (_Float16*)take((size_t)NGRAPH * HDIM * 2);
    float*    HID  = (float*)take((size_t)NGRAPH * HDIM * 4);

    auto cdiv = [](long a, long b) { return (unsigned)((a + b - 1) / b); };
    const long NH   = (long)N_NODES * HDIM;
    const long ETOT = (long)N_EDGES + N_NODES;

    // x -> f16 (row-major, lda = 64 for layer 1)
    cvt_f16_k<<<cdiv((long)N_NODES * F_IN, 256), 256, 0, stream>>>(X, X16, (long)N_NODES * F_IN);

    auto run_layer = [&](int K, const float* W, const float* as, const float* ad,
                         const float* bias, const float* gm, const float* bt,
                         const float* mu, const float* vr) {
        int nch = K / 32;
        pack_w_k<<<cdiv((long)nch * 8 * 512, 256), 256, 0, stream>>>(W, WP, nch);
        if (nch == 2)
            wmma_gemm_k<2><<<cdiv(N_NODES, 128), 256, 0, stream>>>(X16, WP, H, K, N_NODES);
        else
            wmma_gemm_k<4><<<cdiv(N_NODES, 128), 256, 0, stream>>>(X16, WP, H, K, N_NODES);
        row_dot2_k<<<cdiv(N_NODES, 8), 256, 0, stream>>>(H, as, ad, S1, S2, N_NODES);
        init_seg_k<<<cdiv(NH, 256), 256, 0, stream>>>(MM, DEN, AGG, N_NODES, NH);
        edge_max_k<<<cdiv(ETOT, 256), 256, 0, stream>>>(EI, S1, S2, MM, N_EDGES, N_NODES);
        edge_expsum_k<<<cdiv(ETOT, 256), 256, 0, stream>>>(EI, S1, S2, MM, DEN, COEF, N_EDGES, N_NODES);
        scatter_k<<<cdiv(ETOT, SC_EDGES), 128, 0, stream>>>(EI, COEF, DEN, H, AGG, N_EDGES, N_NODES);
        postact_k<<<cdiv(NH, 256), 256, 0, stream>>>(AGG, bias, gm, bt, mu, vr, H, X16, NH);
    };

    run_layer(F_IN, W1, as1, ad1, b1, g1, be1, m1, v1);
    run_layer(HDIM, W2, as2, ad2, b2, g2, be2, m2, v2);
    run_layer(HDIM, W3, as3, ad3, b3, g3, be3, m3, v3);

    // global mean pool
    zero_pool_k<<<cdiv((long)NGRAPH * HDIM, 256), 256, 0, stream>>>(POOL, CNT);
    pool_sum_k<<<cdiv(N_NODES, PL_NODES), 128, 0, stream>>>(BATCH, H, POOL, CNT, N_NODES);
    pool_div_k<<<cdiv((long)NGRAPH * HDIM, 256), 256, 0, stream>>>(POOL, CNT, P16);

    // MLP head: pooled @ lw1 (WMMA) + relu, then matvec
    pack_w_k<<<cdiv((long)4 * 8 * 512, 256), 256, 0, stream>>>(lw1, WP, 4);
    wmma_gemm_k<4><<<cdiv(NGRAPH, 128), 256, 0, stream>>>(P16, WP, HID, HDIM, NGRAPH);
    hid_act_k<<<cdiv((long)NGRAPH * HDIM, 256), 256, 0, stream>>>(HID, lb1);
    final_head_k<<<1, 256, 0, stream>>>(HID, lw2, lb2, (float*)d_out);
}
